// DeltaNet_31877247271493
// MI455X (gfx1250) — matrix operand order repaired and checked by reading, not compiled, not run on hardware
//
#include <hip/hip_runtime.h>

// DeltaNet chunkwise forward, MI455X (gfx1250, wave32).
// 128 workgroups = 32 (b*h) x 4 (dv blocks of 32), 1 wave each.
// WMMA f16 (f32 accum) for all matmuls; operands staged in LDS so every
// fragment gather is contiguous. Next chunk's v/beta are double-buffered
// into LDS by the Tensor Data Mover (tensor_load_to_lds, TENSORcnt).

#define DKk   128
#define DVv   128
#define CHc   32
#define VBv   32
#define NVB   4
#define LSEQ  4096
#define NCH   (LSEQ / CHc)
#define BHn   32

#define LDH   136   // padded ld (halves) for 32x128 f16 arrays (row = 272B, 16B-mult)
#define LDS32 132   // padded ld (floats) for SsT32
#define LDT   34    // padded ld (floats) for Ts (even -> float2-aligned)
#define LDQ   40    // padded ld (halves) for 32x32 f16 arrays (row = 80B, 16B-mult)

typedef __attribute__((ext_vector_type(16))) _Float16 v16h;
typedef __attribute__((ext_vector_type(8)))  _Float16 v8h;
typedef __attribute__((ext_vector_type(4)))  _Float16 v4h;
typedef __attribute__((ext_vector_type(8)))  float    v8f;
typedef unsigned int __attribute__((ext_vector_type(4))) u32x4;
typedef int          __attribute__((ext_vector_type(4))) i32x4;
typedef int          __attribute__((ext_vector_type(8))) i32x8;

static __device__ __forceinline__ v8f wmma16(v16h a, v16h b, v8f c) {
    return __builtin_amdgcn_wmma_f32_16x16x32_f16(false, a, false, b, (short)0, c, false, false);
}

// ---- TDM: async 2-D f32 tile load Global -> LDS (D# per ISA 08_async_tensor) ----
static __device__ __forceinline__ void tdm_load_2d_f32(
    const float* gsrc, unsigned lds_addr,
    unsigned tensor_d0, unsigned tensor_d1, unsigned stride_d0,
    unsigned tile_d0, unsigned tile_d1)
{
    const unsigned long long ga = (unsigned long long)(const void*)gsrc;
    u32x4 g0;
    g0[0] = 1u;                                   // count=1, user descriptor
    g0[1] = lds_addr;                             // lds_addr @ [63:32]
    g0[2] = (unsigned)(ga & 0xFFFFFFFFu);         // global_addr[31:0] @ [95:64]
    g0[3] = (unsigned)((ga >> 32) & 0x1FFFFFFu)   // global_addr[56:32] @ [120:96]
          | (2u << 30);                           // type=2 ("image") @ [127:126]
    i32x8 g1;
    g1[0] = (int)(2u << 16);                      // wg_mask=0, data_size=2 (4B)
    g1[1] = (int)((tensor_d0 & 0xFFFFu) << 16);   // tensor_dim0[15:0] @ [63:48]
    g1[2] = (int)(((tensor_d0 >> 16) & 0xFFFFu)   // tensor_dim0[31:16]
          | ((tensor_d1 & 0xFFFFu) << 16));       // tensor_dim1[15:0]
    g1[3] = (int)(((tensor_d1 >> 16) & 0xFFFFu)
          | ((tile_d0 & 0xFFFFu) << 16));         // tile_dim0 @ [127:112]
    g1[4] = (int)(tile_d1 & 0xFFFFu);             // tile_dim1 @ [143:128]; tile_dim2=0
    g1[5] = (int)stride_d0;                       // tensor_dim0_stride[31:0]
    g1[6] = 0;                                    // stride_d0[47:32], stride_d1[15:0]
    g1[7] = 0;
    const i32x4 gz4 = {0, 0, 0, 0};               // groups 2/3: unused (2-D tile)
    const i32x8 gz8 = {0, 0, 0, 0, 0, 0, 0, 0};   // trailing group (clang-23 6-arg form)
    __builtin_amdgcn_tensor_load_to_lds(g0, g1, gz4, gz4, gz8, 0);
}

// ---- fragment gathers (ISA 7.12.2 layouts, wave32) ----
static __device__ __forceinline__ v16h frag_a16(const _Float16* M, int ld, int m0, int k0) {
    const int l = threadIdx.x & 31;
    const _Float16* p = M + (m0 + (l & 15)) * ld + k0 + (l >> 4) * 8;
    union { v16h h; uint4 q[2]; } r;
    r.q[0] = *(const uint4*)(p);
    r.q[1] = *(const uint4*)(p + 16);
    return r.h;
}
static __device__ __forceinline__ v16h frag_bt16(const _Float16* M, int ld, int k0, int n0) {
    const int l = threadIdx.x & 31;
    const _Float16* p = M + (n0 + (l & 15)) * ld + k0 + (l >> 4) * 8;
    union { v16h h; uint4 q[2]; } r;
    r.q[0] = *(const uint4*)(p);
    r.q[1] = *(const uint4*)(p + 16);
    return r.h;
}
// A gathered transposed: A(m,k) = M[k*ld + m]
static __device__ __forceinline__ v16h frag_at16(const _Float16* M, int ld, int m0, int k0) {
    const int l = threadIdx.x & 31;
    const int hf = l >> 4;
    const int m = m0 + (l & 15);
    v16h a;
#pragma unroll
    for (int v = 0; v < 8; ++v) {
        const int kb = k0 + ((v >> 2) * 16) + hf * 8 + (v & 3) * 2;
        a[2 * v]     = M[kb * ld + m];
        a[2 * v + 1] = M[(kb + 1) * ld + m];
    }
    return a;
}
static __device__ __forceinline__ v16h frag_a_f32(const float* M, int ld, int m0, int k0) {
    const int l = threadIdx.x & 31;
    const int m = m0 + (l & 15);
    const int hf = l >> 4;
    v16h a;
#pragma unroll
    for (int v = 0; v < 8; ++v) {
        const int kb = k0 + ((v >> 2) * 16) + hf * 8 + (v & 3) * 2;
        const float2 t = *(const float2*)(M + m * ld + kb);
        a[2 * v]     = (_Float16)t.x;
        a[2 * v + 1] = (_Float16)t.y;
    }
    return a;
}
static __device__ __forceinline__ v16h frag_a_f32_pad16(const float* M, int ld, int m0, int c0) {
    const int l = threadIdx.x & 31;
    const int hf = l >> 4;
    const int m = m0 + (l & 15);
    v16h a;
#pragma unroll
    for (int v = 0; v < 8; ++v) {
        const int kb = ((v >> 2) * 16) + hf * 8 + (v & 3) * 2;
#pragma unroll
        for (int s2 = 0; s2 < 2; ++s2) {
            const int kk = kb + s2;
            a[2 * v + s2] = (kk < 16) ? (_Float16)M[m * ld + c0 + kk] : (_Float16)0.f;
        }
    }
    return a;
}
static __device__ __forceinline__ v16h frag_b_f32_pad16(const float* M, int ld, int k0, int n0) {
    const int l = threadIdx.x & 31;
    const int hf = l >> 4;
    const int n = n0 + (l & 15);
    v16h b;
#pragma unroll
    for (int i = 0; i < 16; ++i) {
        const int kk = (i < 8) ? (hf * 8 + i) : (16 + hf * 8 + (i - 8));
        b[i] = (kk < 16) ? (_Float16)M[(k0 + kk) * ld + n] : (_Float16)0.f;
    }
    return b;
}
static __device__ __forceinline__ v16h frag_scale(v16h a, _Float16 s) {
    v16h r;
#pragma unroll
    for (int i = 0; i < 16; ++i) r[i] = a[i] * s;
    return r;
}
static __device__ __forceinline__ void frag_c_store_f32(float* M, int ld, int m0, int n0, v8f c) {
    const int l = threadIdx.x & 31;
    const int n = n0 + (l & 15);
    const int mb = m0 + 8 * (l >> 4);
#pragma unroll
    for (int r = 0; r < 8; ++r) M[(mb + r) * ld + n] = c[r];
}

__global__ __launch_bounds__(32)
void deltanet_fwd(const float* __restrict__ q, const float* __restrict__ k,
                  const float* __restrict__ v, const float* __restrict__ beta,
                  float* __restrict__ out, float* __restrict__ Sout)
{
    __shared__ __attribute__((aligned(16))) _Float16 qn16[CHc * LDH];    // q * rsqrt-norm
    __shared__ __attribute__((aligned(16))) _Float16 kn16[CHc * LDH];    // k * rsqrt-norm
    __shared__ __attribute__((aligned(16))) _Float16 SsT16[VBv * LDH];   // f16 mirror of S (dv-major)
    __shared__ __attribute__((aligned(16))) float    SsT32[VBv * LDS32]; // f32 master S (dv-major)
    __shared__ __attribute__((aligned(16))) _Float16 vsT16[VBv * LDQ];   // v*beta (dv-major)
    __shared__ __attribute__((aligned(16))) float    Ts[CHc * LDT];      // f32 T / o staging
    __shared__ __attribute__((aligned(16))) _Float16 attns16[CHc * LDQ]; // tril(q k^T)
    __shared__ __attribute__((aligned(16))) _Float16 usT16[VBv * LDQ];   // tmpT / Y^T / u^T
    __shared__ __attribute__((aligned(16))) _Float16 betas16[CHc];
    __shared__ __attribute__((aligned(16))) float    vstage[CHc * VBv];  // TDM staging: raw v tile
    __shared__ __attribute__((aligned(16))) float    betast[CHc];        // TDM staging: raw beta

    const int lane = threadIdx.x;
    const int bh = blockIdx.x / NVB;
    const int vb = blockIdx.x % NVB;

    for (int idx = lane; idx < VBv * LDS32; idx += 32) SsT32[idx] = 0.0f;
    for (int idx = lane; idx < VBv * LDH; idx += 32) SsT16[idx] = (_Float16)0.f;
    __syncthreads();

    // prologue: TDM-stage chunk 0's v tile and beta row
    {
        const float* vg0 = v + (size_t)bh * LSEQ * DVv + vb * VBv;
        const float* bg0 = beta + (size_t)bh * LSEQ;
        tdm_load_2d_f32(vg0, (unsigned)(unsigned long long)(const void*)vstage,
                        DVv, CHc, DVv, VBv, CHc);
        tdm_load_2d_f32(bg0, (unsigned)(unsigned long long)(const void*)betast,
                        CHc, 1, CHc, CHc, 1);
    }

    for (int ch = 0; ch < NCH; ++ch) {
        const float* qg = q + ((size_t)bh * LSEQ + ch * CHc) * DKk;
        const float* kg = k + ((size_t)bh * LSEQ + ch * CHc) * DKk;

        if (ch + 1 < NCH) {   // L2 prefetch of next chunk's q/k (global_prefetch_b8)
            const float* qn = qg + CHc * DKk;
            const float* kn = kg + CHc * DKk;
#pragma unroll
            for (int i = 0; i < 4; ++i) {
                __builtin_prefetch(qn + (i * 32 + lane) * 32, 0, 1);
                __builtin_prefetch(kn + (i * 32 + lane) * 32, 0, 1);
            }
        }

        // ---- wait for TDM-staged v/beta, then fused load + l2norm pass ----
        __builtin_amdgcn_s_wait_tensorcnt(0);
        const float bl = betast[lane];
        betas16[lane] = (_Float16)bl;
        for (int r = 0; r < CHc; ++r) {
            const float4 q4 = *(const float4*)(qg + r * DKk + lane * 4);
            const float4 k4 = *(const float4*)(kg + r * DKk + lane * 4);
            float sq = q4.x * q4.x + q4.y * q4.y + q4.z * q4.z + q4.w * q4.w;
            float sk = k4.x * k4.x + k4.y * k4.y + k4.z * k4.z + k4.w * k4.w;
#pragma unroll
            for (int m2 = 16; m2 >= 1; m2 >>= 1) {
                sq += __shfl_xor(sq, m2, 32);
                sk += __shfl_xor(sk, m2, 32);
            }
            const float rq = rsqrtf(sq + 1e-6f);
            const float rk = rsqrtf(sk + 1e-6f);
            v4h qh = { (_Float16)(q4.x * rq), (_Float16)(q4.y * rq),
                       (_Float16)(q4.z * rq), (_Float16)(q4.w * rq) };
            v4h kh = { (_Float16)(k4.x * rk), (_Float16)(k4.y * rk),
                       (_Float16)(k4.z * rk), (_Float16)(k4.w * rk) };
            *(v4h*)(qn16 + r * LDH + lane * 4) = qh;
            *(v4h*)(kn16 + r * LDH + lane * 4) = kh;
            const float bb = __shfl(bl, r, 32);
            vsT16[lane * LDQ + r] = (_Float16)(vstage[r * VBv + lane] * bb);
        }
        __syncthreads();

        // kick off TDM for next chunk's v/beta (overlaps all compute below)
        if (ch + 1 < NCH) {
            const float* vgn = v + ((size_t)bh * LSEQ + (ch + 1) * CHc) * DVv + vb * VBv;
            const float* bgn = beta + (size_t)bh * LSEQ + (ch + 1) * CHc;
            tdm_load_2d_f32(vgn, (unsigned)(unsigned long long)(const void*)vstage,
                            DVv, CHc, DVv, VBv, CHc);
            tdm_load_2d_f32(bgn, (unsigned)(unsigned long long)(const void*)betast,
                            CHc, 1, CHc, CHc, 1);
        }

        const _Float16 bm0 = betas16[lane & 15];
        const _Float16 bm1 = betas16[16 + (lane & 15)];

        // ---- fused: Amat = (beta*k)@k^T and attn = q@k^T (share B frags) ----
        {
            v8f cA[2][2] = {{{}, {}}, {{}, {}}};
            v8f cQ[2][2] = {{{}, {}}, {{}, {}}};
#pragma unroll
            for (int kb2 = 0; kb2 < 4; ++kb2) {
                const v16h bt0 = frag_bt16(kn16, LDH, kb2 * 32, 0);
                const v16h bt1 = frag_bt16(kn16, LDH, kb2 * 32, 16);
                const v16h ka0 = frag_scale(frag_a16(kn16, LDH, 0, kb2 * 32), bm0);
                const v16h ka1 = frag_scale(frag_a16(kn16, LDH, 16, kb2 * 32), bm1);
                const v16h qa0 = frag_a16(qn16, LDH, 0, kb2 * 32);
                const v16h qa1 = frag_a16(qn16, LDH, 16, kb2 * 32);
                cA[0][0] = wmma16(ka0, bt0, cA[0][0]);
                cA[0][1] = wmma16(ka0, bt1, cA[0][1]);
                cA[1][0] = wmma16(ka1, bt0, cA[1][0]);
                cA[1][1] = wmma16(ka1, bt1, cA[1][1]);
                cQ[0][0] = wmma16(qa0, bt0, cQ[0][0]);
                cQ[0][1] = wmma16(qa0, bt1, cQ[0][1]);
                cQ[1][0] = wmma16(qa1, bt0, cQ[1][0]);
                cQ[1][1] = wmma16(qa1, bt1, cQ[1][1]);
            }
#pragma unroll
            for (int tm = 0; tm < 2; ++tm)
#pragma unroll
                for (int tn = 0; tn < 2; ++tn) {
                    const int n = tn * 16 + (lane & 15);
                    const int mb = tm * 16 + 8 * (lane >> 4);
#pragma unroll
                    for (int r = 0; r < 8; ++r) {
                        const int m = mb + r;
                        Ts[m * LDT + n] = (m > n) ? -cA[tm][tn][r] : 0.0f;
                        attns16[m * LDQ + n] = (m >= n) ? (_Float16)cQ[tm][tn][r]
                                                        : (_Float16)0.f;
                    }
                }
        }
        __syncthreads();

        // ---- blocked (I - L)^-1: two 16x16 solves in parallel, T21 via WMMA ----
        {
            const int bcol = lane & 15;
            const int bs = (lane >> 4) * 16;
            for (int i = 1; i < 16; ++i) {
                float acc = 0.f;
                for (int t = 0; t < i; ++t)
                    acc += Ts[(bs + i) * LDT + (bs + t)] * Ts[(bs + t) * LDT + (bs + bcol)];
                __syncthreads();
                if (bcol < i) Ts[(bs + i) * LDT + (bs + bcol)] += acc;
                __syncthreads();
            }
            Ts[lane * LDT + lane] += 1.0f;   // diag blocks now hold T11, T22
        }
        __syncthreads();
        {   // tmp = N21 @ T11 (stored transposed + zero-padded into usT16)
            v16h a = frag_a_f32_pad16(Ts, LDT, 16, 0);
            v16h b = frag_b_f32_pad16(Ts, LDT, 0, 0);
            v8f c = {};
            c = wmma16(a, b, c);
            const int n = lane & 15;
            const int mb = 8 * (lane >> 4);
#pragma unroll
            for (int r = 0; r < 8; ++r) {
                usT16[n * LDQ + (mb + r)] = (_Float16)c[r];
                usT16[n * LDQ + 16 + (mb + r)] = (_Float16)0.f;
            }
        }
        __syncthreads();
        {   // T21 = T22 @ tmp -> Ts block (1,0)
            v16h a = frag_a_f32_pad16(Ts, LDT, 16, 16);
            v16h b = frag_bt16(usT16, LDQ, 0, 0);
            v8f c = {};
            c = wmma16(a, b, c);
            frag_c_store_f32(Ts, LDT, 16, 0, c);
        }
        __syncthreads();

        // ---- X = (beta*k) @ S ;  Y = v*beta - X (stored Y^T into usT16) ----
        {
            v8f x[2][2] = {{{}, {}}, {{}, {}}};
#pragma unroll
            for (int kb2 = 0; kb2 < 4; ++kb2) {
                const v16h a0 = frag_scale(frag_a16(kn16, LDH, 0, kb2 * 32), bm0);
                const v16h a1 = frag_scale(frag_a16(kn16, LDH, 16, kb2 * 32), bm1);
                const v16h b0 = frag_bt16(SsT16, LDH, kb2 * 32, 0);
                const v16h b1 = frag_bt16(SsT16, LDH, kb2 * 32, 16);
                x[0][0] = wmma16(a0, b0, x[0][0]);
                x[0][1] = wmma16(a0, b1, x[0][1]);
                x[1][0] = wmma16(a1, b0, x[1][0]);
                x[1][1] = wmma16(a1, b1, x[1][1]);
            }
#pragma unroll
            for (int tm = 0; tm < 2; ++tm)
#pragma unroll
                for (int tn = 0; tn < 2; ++tn) {
                    const int n = tn * 16 + (lane & 15);
                    const int mb = tm * 16 + 8 * (lane >> 4);
                    const v8h vh = *(const v8h*)(vsT16 + n * LDQ + mb);
                    v8h yh;
#pragma unroll
                    for (int r = 0; r < 8; ++r)
                        yh[r] = (_Float16)((float)vh[r] - x[tm][tn][r]);
                    *(v8h*)(usT16 + n * LDQ + mb) = yh;
                }
        }
        __syncthreads();

        // ---- u = T @ Y (then overwrite usT16 with u^T) ----
        {
            const v16h ta0 = frag_a_f32(Ts, LDT, 0, 0);
            const v16h ta1 = frag_a_f32(Ts, LDT, 16, 0);
            const v16h yb0 = frag_bt16(usT16, LDQ, 0, 0);
            const v16h yb1 = frag_bt16(usT16, LDQ, 0, 16);
            v8f u00 = {}, u01 = {}, u10 = {}, u11 = {};
            u00 = wmma16(ta0, yb0, u00);
            u01 = wmma16(ta0, yb1, u01);
            u10 = wmma16(ta1, yb0, u10);
            u11 = wmma16(ta1, yb1, u11);
            v8f ua[2][2] = {{u00, u01}, {u10, u11}};
#pragma unroll
            for (int tm = 0; tm < 2; ++tm)
#pragma unroll
                for (int tn = 0; tn < 2; ++tn) {
                    const int n = tn * 16 + (lane & 15);
                    const int mb = tm * 16 + 8 * (lane >> 4);
                    v8h hh;
#pragma unroll
                    for (int r = 0; r < 8; ++r) hh[r] = (_Float16)ua[tm][tn][r];
                    *(v8h*)(usT16 + n * LDQ + mb) = hh;
                }
        }
        __syncthreads();

        // ---- o = q@S + attn@u (staged in Ts, coalesced store) ----
        const v16h ub0 = frag_bt16(usT16, LDQ, 0, 0);
        const v16h ub1 = frag_bt16(usT16, LDQ, 0, 16);
        {
            v8f o[2][2] = {{{}, {}}, {{}, {}}};
#pragma unroll
            for (int kb2 = 0; kb2 < 4; ++kb2) {
                const v16h a0 = frag_a16(qn16, LDH, 0, kb2 * 32);
                const v16h a1 = frag_a16(qn16, LDH, 16, kb2 * 32);
                const v16h b0 = frag_bt16(SsT16, LDH, kb2 * 32, 0);
                const v16h b1 = frag_bt16(SsT16, LDH, kb2 * 32, 16);
                o[0][0] = wmma16(a0, b0, o[0][0]);
                o[0][1] = wmma16(a0, b1, o[0][1]);
                o[1][0] = wmma16(a1, b0, o[1][0]);
                o[1][1] = wmma16(a1, b1, o[1][1]);
            }
            const v16h aA0 = frag_a16(attns16, LDQ, 0, 0);
            const v16h aA1 = frag_a16(attns16, LDQ, 16, 0);
            o[0][0] = wmma16(aA0, ub0, o[0][0]);
            o[0][1] = wmma16(aA0, ub1, o[0][1]);
            o[1][0] = wmma16(aA1, ub0, o[1][0]);
            o[1][1] = wmma16(aA1, ub1, o[1][1]);
#pragma unroll
            for (int tm = 0; tm < 2; ++tm)
#pragma unroll
                for (int tn = 0; tn < 2; ++tn)
                    frag_c_store_f32(Ts, LDT, tm * 16, tn * 16, o[tm][tn]);
        }
        __syncthreads();
        {
            float* og = out + ((size_t)bh * LSEQ + ch * CHc) * DVv + vb * VBv;
            for (int j = 0; j < CHc; ++j)
                og[(size_t)j * DVv + lane] = Ts[j * LDT + lane];
        }

        // ---- S += k^T @ u  (f32 master dv-major + f16 mirror); reuses ub0/ub1 ----
#pragma unroll
        for (int tm = 0; tm < 8; ++tm) {
            const v16h a = frag_at16(kn16, LDH, tm * 16, 0);
#pragma unroll
            for (int tn = 0; tn < 2; ++tn) {
                const int n = tn * 16 + (lane & 15);
                const int mb = tm * 16 + 8 * (lane >> 4);
                float* p = SsT32 + n * LDS32 + mb;
                union { v8f f; uint4 qq[2]; } cc;
                cc.qq[0] = *(const uint4*)(p);
                cc.qq[1] = *(const uint4*)(p + 4);
                cc.f = wmma16(a, tn ? ub1 : ub0, cc.f);
                *(uint4*)(p) = cc.qq[0];
                *(uint4*)(p + 4) = cc.qq[1];
                v8h hh;
#pragma unroll
                for (int r = 0; r < 8; ++r) hh[r] = (_Float16)cc.f[r];
                *(v8h*)(SsT16 + n * LDH + mb) = hh;
            }
        }
        __syncthreads();
    }

    // ---- final state: SsT32 (dv-major) -> Sout (dk-major) ----
    {
        float* sg = Sout + (size_t)bh * DKk * DVv + vb * VBv;
        for (int d = 0; d < DKk; ++d)
            sg[(size_t)d * DVv + lane] = SsT32[lane * LDS32 + d];
    }
}

extern "C" void kernel_launch(void* const* d_in, const int* in_sizes, int n_in,
                              void* d_out, int out_size, void* d_ws, size_t ws_size,
                              hipStream_t stream) {
    const float* q    = (const float*)d_in[0];
    const float* k    = (const float*)d_in[1];
    const float* v    = (const float*)d_in[2];
    const float* beta = (const float*)d_in[3];
    float* out  = (float*)d_out;
    float* Sout = out + (size_t)BHn * LSEQ * DVv;   // out then S, concatenated flat
    dim3 grid(BHn * NVB), block(32);
    hipLaunchKernelGGL(deltanet_fwd, grid, block, 0, stream, q, k, v, beta, out, Sout);
}
